// KANConvTranspose2d_16509854286317
// MI455X (gfx1250) — matrix operand order repaired and checked by reading, not compiled, not run on hardware
//
#include <hip/hip_runtime.h>

typedef __attribute__((ext_vector_type(16))) _Float16 v16h;
typedef __attribute__((ext_vector_type(8)))  _Float16 v8h;
typedef __attribute__((ext_vector_type(8)))  float    v8f;

#define NB    16
#define CIN   256
#define HH    32
#define WW    32
#define COUT  128
#define NG    8                    // GRID_SIZE + SPLINE_ORDER
#define NF    9                    // silu + 8 spline bases
#define MTOT  (NB*HH*WW)           // 16384
#define NTOT  (COUT*4)             // 512
#define KTOT  (NF*CIN)             // 2304

// ---------------------------------------------------------------------------
// Cox-de Boor order-3 B-spline bases on the uniform knot grid
// t[i] = 0.4*i - 2.2, i = 0..11  (12 knots -> 8 order-3 bases)
// ---------------------------------------------------------------------------
__device__ __forceinline__ void bspline8(float x, float* out) {
    float b[11];
#pragma unroll
    for (int i = 0; i < 11; ++i) {
        float t0 = 0.4f * i - 2.2f;
        float t1 = t0 + 0.4f;
        b[i] = (x >= t0 && x < t1) ? 1.0f : 0.0f;
    }
#pragma unroll
    for (int k = 1; k <= 3; ++k) {
        float inv = 1.0f / (0.4f * (float)k);
#pragma unroll
        for (int i = 0; i < 11 - k; ++i) {
            float ti   = 0.4f * i - 2.2f;
            float tik1 = 0.4f * (i + k + 1) - 2.2f;
            b[i] = (x - ti) * inv * b[i] + (tik1 - x) * inv * b[i + 1];
        }
    }
#pragma unroll
    for (int i = 0; i < 8; ++i) out[i] = b[i];
}

// ---------------------------------------------------------------------------
// Kernel 1: feature expansion  F[row][k], row = b*1024+h*32+w, k = j*256+c
// ---------------------------------------------------------------------------
__global__ __launch_bounds__(256) void kan_features(const float* __restrict__ x,
                                                    _Float16* __restrict__ F) {
    int bh = blockIdx.x;            // b*32 + h
    int b  = bh >> 5;
    int h  = bh & 31;
    int c  = threadIdx.x;           // 0..255
    const float* xp = x + (((size_t)(b * CIN + c) * HH + h) * WW);
    size_t rowBase = (size_t)(b * HH * WW + h * WW) * KTOT;
#pragma unroll 4
    for (int w = 0; w < WW; ++w) {
        float xv = xp[w];
        float s  = xv / (1.0f + __expf(-xv));      // silu
        float bas[8];
        bspline8(xv, bas);
        _Float16* fp = F + rowBase + (size_t)w * KTOT + c;
        fp[0] = (_Float16)s;
#pragma unroll
        for (int g = 0; g < 8; ++g) fp[(size_t)(g + 1) * CIN] = (_Float16)bas[g];
    }
}

// ---------------------------------------------------------------------------
// Kernel 2: fuse weights into fp16 W2[n][k] (n = o*4 + m*2 + nn, spatially
// flipped), and compute the f32 bias Z[n] = total0[o] - active0[o,m,nn].
// ---------------------------------------------------------------------------
__global__ __launch_bounds__(256) void kan_weights(const float* __restrict__ bw,
                                                   const float* __restrict__ sw,
                                                   _Float16* __restrict__ W2,
                                                   float* __restrict__ Z) {
    __shared__ float red[4][256];
    int o = blockIdx.x;
    int c = threadIdx.x;
    float b0[8];
    bspline8(0.0f, b0);

#pragma unroll
    for (int p = 0; p < 2; ++p) {
#pragma unroll
        for (int q = 0; q < 2; ++q) {
            size_t widx = (((size_t)o * CIN + c) * 2 + p) * 2 + q;
            const float* sp = sw + widx * NG;
            float bv = bw[widx];
            int nIdx = o * 4 + (1 - p) * 2 + (1 - q);   // spatial flip
            _Float16* wp = W2 + (size_t)nIdx * KTOT + c;
            wp[0] = (_Float16)bv;
            float acc = 0.0f;
#pragma unroll
            for (int g = 0; g < NG; ++g) {
                float v = sp[g];
                wp[(size_t)(g + 1) * CIN] = (_Float16)v;
                acc += v * b0[g];
            }
            red[p * 2 + q][c] = acc;
        }
    }
    __syncthreads();
    if (c < 4) {
        float s = 0.0f;
        for (int i = 0; i < 256; ++i) s += red[c][i];
        red[c][0] = s;
    }
    __syncthreads();
    if (c == 0) {
        float S00 = red[0][0], S01 = red[1][0], S10 = red[2][0], S11 = red[3][0];
        float total = S00 + S01 + S10 + S11;
        Z[o * 4 + 0] = total - S11;   // m=0,nn=0
        Z[o * 4 + 1] = total - S10;   // m=0,nn=1
        Z[o * 4 + 2] = total - S01;   // m=1,nn=0
        Z[o * 4 + 3] = total - S00;   // m=1,nn=1
    }
}

// ---------------------------------------------------------------------------
// Kernel 3: WMMA GEMM, double-buffered with GLOBAL_LOAD_ASYNC_TO_LDS_B128.
// Block tile 256x128, 8 waves in 4x2 grid, 64x64 per wave -> 16 WMMA/K-step.
// ---------------------------------------------------------------------------
#define TM 256
#define TN 128
#define KS 32
#define LDA 40      // halves per LDS row: 32 data + 8 pad -> 80B, 16B aligned

__device__ __forceinline__ void async_ld16(unsigned lds_addr,
                                           unsigned long long gaddr) {
    // GLOBAL_LOAD_ASYNC_TO_LDS_B128: VDST = LDS byte address, VADDR = 64b addr
    asm volatile("global_load_async_to_lds_b128 %0, %1, off"
                 :: "v"(lds_addr), "v"(gaddr) : "memory");
}

__global__ __launch_bounds__(256) void kan_gemm(const _Float16* __restrict__ F,
                                                const _Float16* __restrict__ W2,
                                                const float* __restrict__ Z,
                                                float* __restrict__ out) {
    __shared__ _Float16 As[2][TM * LDA];   // 2 x 20 KB
    __shared__ _Float16 Bs[2][TN * LDA];   // 2 x 10 KB

    int tid   = threadIdx.x;
    int lane  = tid & 31;
    int wave  = tid >> 5;        // 0..7
    int wm    = wave >> 1;       // 0..3  (64-row strip)
    int wn    = wave & 1;        // 0..1  (64-col strip)
    int tileM = blockIdx.x * TM;
    int tileN = blockIdx.y * TN;
    int l16   = lane & 15;
    int lhi   = lane >> 4;       // lane-group: 0 or 1

    // -------- per-thread staging addresses (6 x 16B chunks per K-slab) -----
    unsigned long long gA[4], gB[2];
    unsigned lA[2][4], lB[2][2];
#pragma unroll
    for (int i = 0; i < 4; ++i) {
        int e = tid + i * 256;              // 0..1023: A chunk id
        int r = e >> 2, c4 = e & 3;
        gA[i] = (unsigned long long)(F + (size_t)(tileM + r) * KTOT + c4 * 8);
        lA[0][i] = (unsigned)(size_t)&As[0][r * LDA + c4 * 8];
        lA[1][i] = (unsigned)(size_t)&As[1][r * LDA + c4 * 8];
    }
#pragma unroll
    for (int i = 0; i < 2; ++i) {
        int e = tid + i * 256;              // 0..511: B chunk id
        int r = e >> 2, c4 = e & 3;
        gB[i] = (unsigned long long)(W2 + (size_t)(tileN + r) * KTOT + c4 * 8);
        lB[0][i] = (unsigned)(size_t)&Bs[0][r * LDA + c4 * 8];
        lB[1][i] = (unsigned)(size_t)&Bs[1][r * LDA + c4 * 8];
    }

    v8f acc[4][4] = {};

    const int NKT = KTOT / KS;              // 72 K-steps

    // prologue: fill buffer 0
#pragma unroll
    for (int i = 0; i < 4; ++i) async_ld16(lA[0][i], gA[i]);
#pragma unroll
    for (int i = 0; i < 2; ++i) async_ld16(lB[0][i], gB[i]);

    for (int kt = 0; kt < NKT; ++kt) {
        int cur = kt & 1;
        if (kt + 1 < NKT) {
            // issue next K-slab into the other buffer (6 more in flight)
            unsigned long long koff = (unsigned long long)((kt + 1) * KS) * 2u;
#pragma unroll
            for (int i = 0; i < 4; ++i) async_ld16(lA[1 - cur][i], gA[i] + koff);
#pragma unroll
            for (int i = 0; i < 2; ++i) async_ld16(lB[1 - cur][i], gB[i] + koff);
            // in-order completion: <=6 outstanding => current slab landed
            asm volatile("s_wait_asynccnt 0x6" ::: "memory");
        } else {
            asm volatile("s_wait_asynccnt 0x0" ::: "memory");
        }
        __syncthreads();

        const _Float16* Ab = As[cur];
        const _Float16* Bb = Bs[cur];

        // ---- fragments per ISA 16-bit A/B lane layouts ----
        v16h afrag[4], bfrag[4];
#pragma unroll
        for (int mi = 0; mi < 4; ++mi) {
            int row = wm * 64 + mi * 16 + l16;
            v8h lo = *(const v8h*)(&Ab[row * LDA + lhi * 8]);        // K=lhi*8..+7
            v8h hi = *(const v8h*)(&Ab[row * LDA + lhi * 8 + 16]);   // K=lhi*8+16..+23
            v16h a;
#pragma unroll
            for (int j = 0; j < 8; ++j) { a[j] = lo[j]; a[j + 8] = hi[j]; }
            afrag[mi] = a;
        }
#pragma unroll
        for (int ni = 0; ni < 4; ++ni) {
            int col = wn * 64 + ni * 16 + l16;
            v8h lo = *(const v8h*)(&Bb[col * LDA + lhi * 8]);        // K=lhi*8..+7
            v8h hi = *(const v8h*)(&Bb[col * LDA + lhi * 8 + 16]);   // K=lhi*8+16..+23
            v16h b;
#pragma unroll
            for (int j = 0; j < 8; ++j) { b[j] = lo[j]; b[j + 8] = hi[j]; }
            bfrag[ni] = b;
        }

        // ---- 16 WMMAs per K-step ----
#pragma unroll
        for (int mi = 0; mi < 4; ++mi)
#pragma unroll
            for (int ni = 0; ni < 4; ++ni)
                acc[mi][ni] = __builtin_amdgcn_wmma_f32_16x16x32_f16(
                    false, afrag[mi], false, bfrag[ni],
                    (short)0, acc[mi][ni], false, false);

        __syncthreads();   // protects buffer reuse at kt+2
    }

    // ---- epilogue: add bias, scatter into [B,Cout,64,64] ----
#pragma unroll
    for (int ni = 0; ni < 4; ++ni) {
        int n    = tileN + wn * 64 + ni * 16 + l16;
        float bias = Z[n];
        int o  = n >> 2;
        int km = (n >> 1) & 1;
        int kn = n & 1;
#pragma unroll
        for (int mi = 0; mi < 4; ++mi) {
#pragma unroll
            for (int i = 0; i < 8; ++i) {
                int rowg = tileM + wm * 64 + mi * 16 + i + 8 * lhi;
                int b    = rowg >> 10;
                int rem  = rowg & 1023;
                int h    = rem >> 5;
                int w    = rem & 31;
                size_t oi = (((size_t)b * COUT + o) * 64 + (2 * h + km)) * 64
                          + (2 * w + kn);
                out[oi] = acc[mi][ni][i] + bias;
            }
        }
    }
}

// ---------------------------------------------------------------------------
extern "C" void kernel_launch(void* const* d_in, const int* in_sizes, int n_in,
                              void* d_out, int out_size, void* d_ws, size_t ws_size,
                              hipStream_t stream) {
    const float* x  = (const float*)d_in[0];   // [16,256,32,32]
    const float* bw = (const float*)d_in[1];   // [128,256,2,2]
    const float* sw = (const float*)d_in[2];   // [128,256,2,2,8]
    float* out = (float*)d_out;                // [16,128,64,64]

    _Float16* F  = (_Float16*)d_ws;                                    // 75.5 MB
    _Float16* W2 = (_Float16*)((char*)d_ws + (size_t)MTOT * KTOT * 2); // 2.25 MB
    float*    Z  = (float*)((char*)d_ws + (size_t)MTOT * KTOT * 2
                                        + (size_t)NTOT * KTOT * 2);    // 2 KB

    kan_features<<<NB * HH, 256, 0, stream>>>(x, F);
    kan_weights <<<COUT,    256, 0, stream>>>(bw, sw, W2, Z);
    kan_gemm    <<<dim3(MTOT / TM, NTOT / TN), 256, 0, stream>>>(F, W2, Z, out);
}
